// MultiHeadSolitonAttention_81853486727570
// MI455X (gfx1250) — compile-verified, matrix operand order checked by reading, not run on hardware
//
#include <hip/hip_runtime.h>
#include <hip/hip_bf16.h>

typedef __attribute__((ext_vector_type(16))) _Float16 v16h;
typedef __attribute__((ext_vector_type(8)))  _Float16 v8h;
typedef __attribute__((ext_vector_type(4)))  _Float16 v4h;
typedef __attribute__((ext_vector_type(8)))  float    v8f;
typedef __attribute__((ext_vector_type(4)))  float    v4fv;
typedef __attribute__((ext_vector_type(4)))  int      v4i;

#define PDE_DT  0.05f
#define HJ_H    0.05f
#define T_SEQ   2048
#define EMBED   768
#define GD      256
#define MROWS   8192           // B*T = 4*2048

// ---------------------------------------------------------------------------
// gfx1250 async global->LDS copies (ASYNCcnt-tracked), guarded so that either
// toolchain compiles; fallback is the plain load/store copy.
// Builtin signatures (from hipcc diagnostics):
//   b128: (AS1 v4i*, AS3 v4i*, imm offset, imm cpol)
//   b32 : (AS1 int*, AS3 int*, imm offset, imm cpol)
// ---------------------------------------------------------------------------
#if defined(__HIP_DEVICE_COMPILE__) && defined(__gfx1250__) && \
    __has_builtin(__builtin_amdgcn_global_load_async_to_lds_b128) && \
    __has_builtin(__builtin_amdgcn_global_load_async_to_lds_b32) && \
    __has_builtin(__builtin_amdgcn_s_wait_asynccnt)
#define HAVE_ASYNC 1
#else
#define HAVE_ASYNC 0
#endif

// copy 16 bytes global->LDS
__device__ __forceinline__ void cp_g2l_b128(const _Float16* g, _Float16* l) {
#if HAVE_ASYNC
  __builtin_amdgcn_global_load_async_to_lds_b128(
      (__attribute__((address_space(1))) v4i*)(g),
      (__attribute__((address_space(3))) v4i*)(l), 0, 0);
#else
  *(v8h*)l = *(const v8h*)g;
#endif
}

// copy 4 bytes global->LDS
__device__ __forceinline__ void cp_g2l_b32(const float* g, float* l) {
#if HAVE_ASYNC
  __builtin_amdgcn_global_load_async_to_lds_b32(
      (__attribute__((address_space(1))) int*)(g),
      (__attribute__((address_space(3))) int*)(l), 0, 0);
#else
  *l = *g;
#endif
}

__device__ __forceinline__ void async_wait() {
#if HAVE_ASYNC
  __builtin_amdgcn_s_wait_asynccnt(0);
#endif
}

// ---------------------------------------------------------------------------
// WMMA fragment load (wave32, V_WMMA_F32_16X16X32_F16)
// A-fragment (16x32 f16, row = M):
//   lanes 0-15:  halves 0-7 = K0-7,  halves 8-15 = K16-23
//   lanes 16-31: halves 0-7 = K8-15, halves 8-15 = K24-31
// B-fragment is symmetric with row = N.
// ---------------------------------------------------------------------------
__device__ __forceinline__ v16h frag_ld(const _Float16* p0, int stride, int lane) {
  const int r  = lane & 15;
  const int kb = (lane & 16) >> 1;        // 0 or 8
  const _Float16* p = p0 + r * stride + kb;
  v8h lo = *(const v8h*)(p);
  v8h hi = *(const v8h*)(p + 16);
  return __builtin_shufflevector(lo, hi, 0,1,2,3,4,5,6,7,8,9,10,11,12,13,14,15);
}

// ---------------------------------------------------------------------------
// fp32 -> f16 convert (4-wide)
// ---------------------------------------------------------------------------
__global__ void __launch_bounds__(256) cvt_f16_kernel(const float* __restrict__ src,
                                                      _Float16* __restrict__ dst, int n4) {
  int i = blockIdx.x * 256 + threadIdx.x;
  if (i < n4) {
    v4fv v = ((const v4fv*)src)[i];
    ((v4h*)dst)[i] = __builtin_convertvector(v, v4h);
  }
}

// Weight convert + transpose: src fp32 (K x N) -> dst f16 (N x K)
__global__ void __launch_bounds__(256) wt_convert_kernel(const float* __restrict__ src,
                                                         _Float16* __restrict__ dst,
                                                         int K, int N) {
  int i = blockIdx.x * 256 + threadIdx.x;
  if (i < K * N) {
    int k = i / N, n = i - k * N;
    dst[n * K + k] = (_Float16)src[i];
  }
}

// ---------------------------------------------------------------------------
// GEMM: C(M x N) = A(M x K, f16, lda) * Wt(N x K, f16) + bias
// block tile 128x64, 8 waves; each wave owns 4 accumulators in one tile
// column (shared B fragment), K-step 32.  OUT_F16 selects f16 vs f32 C.
// ---------------------------------------------------------------------------
template <bool OUT_F16>
__global__ void __launch_bounds__(256) gemm_wmma_kernel(
    const _Float16* __restrict__ A, int lda,
    const _Float16* __restrict__ Wt,
    const float* __restrict__ bias,
    void* __restrict__ Cv, int ldc, int K) {
  __shared__ __align__(16) _Float16 sA[128 * 32];
  __shared__ __align__(16) _Float16 sB[64 * 32];
  const int bm   = blockIdx.x * 128;
  const int bn   = blockIdx.y * 64;
  const int tid  = threadIdx.x;
  const int wave = tid >> 5, lane = tid & 31;
  const int trb  = wave >> 2;           // base tile row (0..1), +2 per q
  const int tc   = wave & 3;            // tile col (shared by all 4 accs)

  // staging coordinates (b128 chunks)
  const int ar = tid >> 1, ac = (tid & 1) * 16;     // sA: 128 x 32
  const int br = tid >> 2, bc = (tid & 3) * 8;      // sB: 64 x 32

  v8f acc[4] = {{}, {}, {}, {}};

  for (int k0 = 0; k0 < K; k0 += 32) {
    {
      const _Float16* ap = A + (size_t)(bm + ar) * lda + k0 + ac;
      _Float16* al = sA + ar * 32 + ac;
      cp_g2l_b128(ap, al);
      cp_g2l_b128(ap + 8, al + 8);
      cp_g2l_b128(Wt + (size_t)(bn + br) * K + k0 + bc, sB + br * 32 + bc);
    }
    if (k0 + 32 < K)
      __builtin_prefetch(&A[(size_t)(bm + ar) * lda + k0 + 32], 0, 1);
    async_wait();
    __syncthreads();

    v16h b = frag_ld(sB + tc * 16 * 32, 32, lane);
#pragma unroll
    for (int q = 0; q < 4; ++q) {
      v16h a = frag_ld(sA + (trb + 2 * q) * 16 * 32, 32, lane);
      acc[q] = __builtin_amdgcn_wmma_f32_16x16x32_f16(false, a, false, b,
                                                      (short)0, acc[q], false, false);
    }
    __syncthreads();
  }

  // Epilogue: C/D layout -> row M = v + 8*(lane>=16), col N = lane%16
  const int n  = lane & 15;
  const int mo = (lane & 16) >> 1;
  const float bv = bias[bn + tc * 16 + n];
#pragma unroll
  for (int q = 0; q < 4; ++q) {
    const int tr = trb + 2 * q;
    if (OUT_F16) {
      _Float16* Cp = (_Float16*)Cv + (size_t)(bm + tr * 16 + mo) * ldc + (bn + tc * 16 + n);
#pragma unroll
      for (int v = 0; v < 8; ++v) Cp[(size_t)v * ldc] = (_Float16)(acc[q][v] + bv);
    } else {
      float* Cp = (float*)Cv + (size_t)(bm + tr * 16 + mo) * ldc + (bn + tc * 16 + n);
#pragma unroll
      for (int v = 0; v < 8; ++v) Cp[(size_t)v * ldc] = acc[q][v] + bv;
    }
  }
}

// ---------------------------------------------------------------------------
// Causal flash attention, 64-query block / workgroup, head_dim = 64.
// qkv f16 (B, T, 768): q @ col h*64, k @ 256+h*64, v @ 512+h*64.
// O written as f16 (consumed by the out-proj WMMA GEMM).
// ---------------------------------------------------------------------------
__global__ void __launch_bounds__(256) attn_wmma_kernel(
    const _Float16* __restrict__ qkv,   // group base (B, T, 768) f16
    _Float16* __restrict__ O) {         // group out  (B, T, 256) f16
  __shared__ __align__(16) _Float16 sQ [64 * 64];
  __shared__ __align__(16) _Float16 sK [64 * 64];
  __shared__ __align__(16) _Float16 sVt[64 * 64];  // transposed: [dim][key]
  __shared__ __align__(16) _Float16 sP [64 * 64];
  __shared__ float sS[64 * 64];
  __shared__ float sM[64], sL[64], sAlpha[64];

  const int qb = blockIdx.x;           // query block (T/64)
  const int h  = blockIdx.y;           // head (4)
  const int b  = blockIdx.z;           // batch (4)
  const int tid  = threadIdx.x;
  const int wave = tid >> 5, lane = tid & 31;
  const int tr0 = wave >> 2,       tc0 = wave & 3;
  const int tr1 = (wave >> 2) + 2, tc1 = wave & 3;
  const size_t rowB = (size_t)b * T_SEQ * EMBED;
  const int qcol = h * 64, kcol = GD + h * 64, vcol = 2 * GD + h * 64;
  const float scl = 0.125f;            // 1/sqrt(64), applied in softmax

  // stage Q (pure async copy; scale folded into softmax)
  {
    const int m = tid >> 2, c = (tid & 3) * 16;
    const _Float16* qp = qkv + rowB + (size_t)(qb * 64 + m) * EMBED + qcol + c;
    _Float16* ql = sQ + m * 64 + c;
    cp_g2l_b128(qp, ql);
    cp_g2l_b128(qp + 8, ql + 8);
  }
  if (tid < 64) { sM[tid] = -1e30f; sL[tid] = 0.f; }

  v8f o0 = {}, o1 = {};

  for (int j = 0; j <= qb; ++j) {
    __syncthreads();
    {
      // K tile: pure async copy
      const int m = tid >> 2, c = (tid & 3) * 16;
      const _Float16* kp = qkv + rowB + (size_t)(j * 64 + m) * EMBED + kcol + c;
      _Float16* kl = sK + m * 64 + c;
      cp_g2l_b128(kp, kl);
      cp_g2l_b128(kp + 8, kl + 8);
      // V transposed: [dim][key] (data transform -> explicit scatter)
      const int r = tid & 63, db = (tid >> 6) * 16;
      const _Float16* vp = qkv + rowB + (size_t)(j * 64 + r) * EMBED + vcol + db;
      v8h v0 = *(const v8h*)(vp), v1 = *(const v8h*)(vp + 8);
#pragma unroll
      for (int e = 0; e < 8; ++e) {
        sVt[(db + e) * 64 + r]     = v0[e];
        sVt[(db + 8 + e) * 64 + r] = v1[e];
      }
    }
    async_wait();
    __syncthreads();

    // S = Q * K^T  (two K-steps of 32 over head_dim)
    v8f s0 = {}, s1 = {};
#pragma unroll
    for (int ks = 0; ks < 64; ks += 32) {
      v16h aq0 = frag_ld(sQ + tr0 * 16 * 64 + ks, 64, lane);
      v16h bk0 = frag_ld(sK + tc0 * 16 * 64 + ks, 64, lane);
      s0 = __builtin_amdgcn_wmma_f32_16x16x32_f16(false, aq0, false, bk0,
                                                  (short)0, s0, false, false);
      v16h aq1 = frag_ld(sQ + tr1 * 16 * 64 + ks, 64, lane);
      v16h bk1 = frag_ld(sK + tc1 * 16 * 64 + ks, 64, lane);
      s1 = __builtin_amdgcn_wmma_f32_16x16x32_f16(false, aq1, false, bk1,
                                                  (short)0, s1, false, false);
    }
    {
      int n = lane & 15, mo = (lane & 16) >> 1;
#pragma unroll
      for (int v = 0; v < 8; ++v) {
        sS[(tr0 * 16 + mo + v) * 64 + tc0 * 16 + n] = s0[v];
        sS[(tr1 * 16 + mo + v) * 64 + tc1 * 16 + n] = s1[v];
      }
    }
    __syncthreads();

    // online softmax (one thread per query row); scale applied here
    if (tid < 64) {
      const int qg = qb * 64 + tid;
      float mold = sM[tid], mx = mold;
      for (int n = 0; n < 64; ++n)
        if (j * 64 + n <= qg) mx = fmaxf(mx, sS[tid * 64 + n] * scl);
      float alpha = __expf(mold - mx);
      float sum = 0.f;
      for (int n = 0; n < 64; ++n) {
        float p = (j * 64 + n <= qg) ? __expf(sS[tid * 64 + n] * scl - mx) : 0.f;
        sP[tid * 64 + n] = (_Float16)p;
        sum += p;
      }
      sM[tid] = mx;
      sL[tid] = sL[tid] * alpha + sum;
      sAlpha[tid] = alpha;
    }
    __syncthreads();

    // O = alpha*O + P * V
    {
      int mo = (lane & 16) >> 1;
#pragma unroll
      for (int v = 0; v < 8; ++v) {
        o0[v] *= sAlpha[tr0 * 16 + mo + v];
        o1[v] *= sAlpha[tr1 * 16 + mo + v];
      }
    }
#pragma unroll
    for (int ks = 0; ks < 64; ks += 32) {
      v16h ap0 = frag_ld(sP  + tr0 * 16 * 64 + ks, 64, lane);
      v16h bv0 = frag_ld(sVt + tc0 * 16 * 64 + ks, 64, lane);
      o0 = __builtin_amdgcn_wmma_f32_16x16x32_f16(false, ap0, false, bv0,
                                                  (short)0, o0, false, false);
      v16h ap1 = frag_ld(sP  + tr1 * 16 * 64 + ks, 64, lane);
      v16h bv1 = frag_ld(sVt + tc1 * 16 * 64 + ks, 64, lane);
      o1 = __builtin_amdgcn_wmma_f32_16x16x32_f16(false, ap1, false, bv1,
                                                  (short)0, o1, false, false);
    }
  }
  __syncthreads();

  // normalize and write out (f16)
  {
    int n = lane & 15, mo = (lane & 16) >> 1;
    const size_t ob = (size_t)b * T_SEQ * GD;
#pragma unroll
    for (int v = 0; v < 8; ++v) {
      int m0 = tr0 * 16 + mo + v;
      O[ob + (size_t)(qb * 64 + m0) * GD + h * 64 + tc0 * 16 + n] = (_Float16)(o0[v] / sL[m0]);
      int m1 = tr1 * 16 + mo + v;
      O[ob + (size_t)(qb * 64 + m1) * GD + h * 64 + tc1 * 16 + n] = (_Float16)(o1[v] / sL[m1]);
    }
  }
}

// ---------------------------------------------------------------------------
// PDE integrators: one (batch, channel) column of T=2048 in LDS, 3 Euler
// steps fully resident.  fp32 in (async b32 staged), f16 out.
// grid = (256 cols, 4 batches), block = 256.
// ---------------------------------------------------------------------------
__global__ void __launch_bounds__(256) pde_sg_kernel(const float* __restrict__ u,
                                                     _Float16* __restrict__ o, int ld) {
  __shared__ float su[T_SEQ];
  const int col = blockIdx.x, b = blockIdx.y, tid = threadIdx.x;
  const float* base = u + (size_t)b * T_SEQ * ld + col;
  _Float16* ob = o + (size_t)b * T_SEQ * ld + col;
  for (int i = tid; i < T_SEQ; i += 256) cp_g2l_b32(base + (size_t)i * ld, &su[i]);
  float w[8];
#pragma unroll
  for (int i = 0; i < 8; ++i) w[i] = 0.f;
  async_wait();
  __syncthreads();
  for (int s = 0; s < 3; ++s) {
    float un[8];
#pragma unroll
    for (int i = 0; i < 8; ++i) {
      int t = tid * 8 + i;
      float um = (t >= 1) ? su[t - 1] : 0.f;
      float up = (t + 1 < T_SEQ) ? su[t + 1] : 0.f;
      float uc = su[t];
      w[i] += PDE_DT * ((um - 2.f * uc + up) - __sinf(uc));
      un[i] = uc + PDE_DT * w[i];
    }
    __syncthreads();
#pragma unroll
    for (int i = 0; i < 8; ++i) su[tid * 8 + i] = un[i];
    __syncthreads();
  }
  for (int i = tid; i < T_SEQ; i += 256) ob[(size_t)i * ld] = (_Float16)su[i];
}

__global__ void __launch_bounds__(256) pde_kdv_kernel(const float* __restrict__ u,
                                                      _Float16* __restrict__ o, int ld) {
  __shared__ float su[T_SEQ];
  const int col = blockIdx.x, b = blockIdx.y, tid = threadIdx.x;
  const float* base = u + (size_t)b * T_SEQ * ld + col;
  _Float16* ob = o + (size_t)b * T_SEQ * ld + col;
  for (int i = tid; i < T_SEQ; i += 256) cp_g2l_b32(base + (size_t)i * ld, &su[i]);
  async_wait();
  __syncthreads();
  for (int s = 0; s < 3; ++s) {
    float un[8];
#pragma unroll
    for (int i = 0; i < 8; ++i) {
      int t = tid * 8 + i;
      float um1 = (t >= 1) ? su[t - 1] : 0.f;
      float up1 = (t + 1 < T_SEQ) ? su[t + 1] : 0.f;
      float um2 = (t >= 2) ? su[t - 2] : 0.f;
      float up2 = (t + 2 < T_SEQ) ? su[t + 2] : 0.f;
      float uc = su[t];
      float d1 = 0.5f * (up1 - um1);
      float d3 = 0.5f * (up2 - 2.f * up1 + 2.f * um1 - um2);
      un[i] = uc + PDE_DT * (-6.f * uc * d1 - d3);
    }
    __syncthreads();
#pragma unroll
    for (int i = 0; i < 8; ++i) su[tid * 8 + i] = un[i];
    __syncthreads();
  }
  for (int i = tid; i < T_SEQ; i += 256) ob[(size_t)i * ld] = (_Float16)su[i];
}

__global__ void __launch_bounds__(256) pde_hj_kernel(const float* __restrict__ u,
                                                     _Float16* __restrict__ o, int ld) {
  __shared__ float su[T_SEQ];
  __shared__ float sf[T_SEQ];
  const int col = blockIdx.x, b = blockIdx.y, tid = threadIdx.x;
  const float* base = u + (size_t)b * T_SEQ * ld + col;
  _Float16* ob = o + (size_t)b * T_SEQ * ld + col;
  for (int i = tid; i < T_SEQ; i += 256) cp_g2l_b32(base + (size_t)i * ld, &su[i]);
  float w[8];
#pragma unroll
  for (int i = 0; i < 8; ++i) w[i] = 0.f;
  async_wait();
  __syncthreads();
  for (int s = 0; s < 3; ++s) {
#pragma unroll
    for (int i = 0; i < 8; ++i) {
      int t = tid * 8 + i;
      float uc = su[t];
      sf[t] = uc - uc * uc + uc * uc * uc;   // (c0^2 + p u + q u^2) u
    }
    __syncthreads();
    float un[8];
#pragma unroll
    for (int i = 0; i < 8; ++i) {
      int t = tid * 8 + i;
      float fm = (t >= 1) ? sf[t - 1] : 0.f;
      float fp = (t + 1 < T_SEQ) ? sf[t + 1] : 0.f;
      float d2f = fm - 2.f * sf[t] + fp;
      float um1 = (t >= 1) ? su[t - 1] : 0.f;
      float up1 = (t + 1 < T_SEQ) ? su[t + 1] : 0.f;
      float um2 = (t >= 2) ? su[t - 2] : 0.f;
      float up2 = (t + 2 < T_SEQ) ? su[t + 2] : 0.f;
      float d4 = up2 - 4.f * up1 + 6.f * su[t] - 4.f * um1 + um2;
      w[i] += PDE_DT * (d2f - HJ_H * d4);
      un[i] = su[t] + PDE_DT * w[i];
    }
    __syncthreads();
#pragma unroll
    for (int i = 0; i < 8; ++i) su[tid * 8 + i] = un[i];
    __syncthreads();
  }
  for (int i = tid; i < T_SEQ; i += 256) ob[(size_t)i * ld] = (_Float16)su[i];
}

// ---------------------------------------------------------------------------
// Host orchestration
// ---------------------------------------------------------------------------
extern "C" void kernel_launch(void* const* d_in, const int* in_sizes, int n_in,
                              void* d_out, int out_size, void* d_ws, size_t ws_size,
                              hipStream_t stream) {
  (void)in_sizes; (void)n_in; (void)out_size; (void)ws_size;
  const float* x    = (const float*)d_in[0];
  const float* in_w = (const float*)d_in[1];
  const float* in_b = (const float*)d_in[2];
  const float* qkv_w[3] = { (const float*)d_in[3],  (const float*)d_in[7],  (const float*)d_in[11] };
  const float* qkv_b[3] = { (const float*)d_in[4],  (const float*)d_in[8],  (const float*)d_in[12] };
  const float* out_w[3] = { (const float*)d_in[5],  (const float*)d_in[9],  (const float*)d_in[13] };
  const float* out_b[3] = { (const float*)d_in[6],  (const float*)d_in[10], (const float*)d_in[14] };
  const float* fw = (const float*)d_in[15];
  const float* fb = (const float*)d_in[16];
  float* out = (float*)d_out;

  // bump-allocate workspace (256B aligned)
  char* ws = (char*)d_ws;
  auto alloc = [&](size_t bytes) -> void* {
    void* p = (void*)ws;
    ws += (bytes + 255) & ~(size_t)255;
    return p;
  };
  _Float16* xh   = (_Float16*)alloc((size_t)MROWS * EMBED * 2);      // x in f16
  _Float16* h    = (_Float16*)alloc((size_t)MROWS * EMBED * 2);      // input proj (f16)
  _Float16* qkv  = (_Float16*)alloc((size_t)3 * MROWS * EMBED * 2);  // per-group qkv (f16)
  _Float16* att  = (_Float16*)alloc((size_t)3 * MROWS * GD * 2);     // attn outputs (f16)
  float*    ho   = (float*)alloc((size_t)MROWS * EMBED * 4);         // concat, fp32 for PDE
  _Float16* hof  = (_Float16*)alloc((size_t)MROWS * EMBED * 2);      // PDE result (f16)
  _Float16* in_wt = (_Float16*)alloc((size_t)EMBED * EMBED * 2);
  _Float16* qkv_wt[3], *out_wt[3];
  for (int g = 0; g < 3; ++g) {
    qkv_wt[g] = (_Float16*)alloc((size_t)(3 * GD) * GD * 2);         // (768 x 256)
    out_wt[g] = (_Float16*)alloc((size_t)GD * GD * 2);               // (256 x 256)
  }
  _Float16* fw_t = (_Float16*)alloc((size_t)EMBED * EMBED * 2);

  // --- converts: x -> f16, weights -> f16 transposed ---
  cvt_f16_kernel<<<(MROWS * EMBED / 4 + 255) / 256, 256, 0, stream>>>(x, xh, MROWS * EMBED / 4);
  wt_convert_kernel<<<(EMBED * EMBED + 255) / 256, 256, 0, stream>>>(in_w, in_wt, EMBED, EMBED);
  for (int g = 0; g < 3; ++g) {
    wt_convert_kernel<<<(GD * 3 * GD + 255) / 256, 256, 0, stream>>>(qkv_w[g], qkv_wt[g], GD, 3 * GD);
    wt_convert_kernel<<<(GD * GD + 255) / 256, 256, 0, stream>>>(out_w[g], out_wt[g], GD, GD);
  }
  wt_convert_kernel<<<(EMBED * EMBED + 255) / 256, 256, 0, stream>>>(fw, fw_t, EMBED, EMBED);

  // --- h = x @ in_w + in_b  (f16 out) ---
  gemm_wmma_kernel<true><<<dim3(MROWS / 128, EMBED / 64), 256, 0, stream>>>(
      xh, EMBED, in_wt, in_b, h, EMBED, EMBED);

  // --- per group: qkv = h_g @ qkv_w + b  (f16 out) ---
  for (int g = 0; g < 3; ++g)
    gemm_wmma_kernel<true><<<dim3(MROWS / 128, (3 * GD) / 64), 256, 0, stream>>>(
        h + g * GD, EMBED, qkv_wt[g], qkv_b[g],
        qkv + (size_t)g * MROWS * EMBED, EMBED, GD);

  // --- per group: causal flash attention (f16 in/out) ---
  for (int g = 0; g < 3; ++g)
    attn_wmma_kernel<<<dim3(T_SEQ / 64, 4, 4), 256, 0, stream>>>(
        qkv + (size_t)g * MROWS * EMBED, att + (size_t)g * MROWS * GD);

  // --- per group: out-proj into concat buffer ho (fp32 for PDE) ---
  for (int g = 0; g < 3; ++g)
    gemm_wmma_kernel<false><<<dim3(MROWS / 128, GD / 64), 256, 0, stream>>>(
        att + (size_t)g * MROWS * GD, GD, out_wt[g], out_b[g],
        ho + g * GD, EMBED, GD);

  // --- PDE integrators: fp32 in, f16 out for the final GEMM ---
  pde_sg_kernel <<<dim3(GD, 4), 256, 0, stream>>>(ho + 0 * GD, hof + 0 * GD, EMBED);
  pde_kdv_kernel<<<dim3(GD, 4), 256, 0, stream>>>(ho + 1 * GD, hof + 1 * GD, EMBED);
  pde_hj_kernel <<<dim3(GD, 4), 256, 0, stream>>>(ho + 2 * GD, hof + 2 * GD, EMBED);

  // --- out = hof @ fw + fb  (fp32 out) ---
  gemm_wmma_kernel<false><<<dim3(MROWS / 128, EMBED / 64), 256, 0, stream>>>(
      hof, EMBED, fw_t, fb, out, EMBED, EMBED);
}
